// EnergyBasedModel_58136677319270
// MI455X (gfx1250) — matrix operand order repaired and checked by reading, not compiled, not run on hardware
//
#include <hip/hip_runtime.h>
#include <hip/hip_bf16.h>
#include <stdint.h>

// ---------------------------------------------------------------------------
// Equilibrium-propagation negative phase on MI455X (gfx1250).
//   - bf16 WMMA (v_wmma_f32_16x16x32_bf16), f32 state math
//   - FFX = sig(x)@W0 + b0 hoisted out of the 20-step loop (x is clamped)
//   - sigmoid(state) maintained as bf16 global buffers, produced by the
//     previous update's epilogue -> GEMM K-loop is pure copies + WMMA
//   - weights pre-swizzled in GLOBAL memory into the WMMA B-fragment layout
//   - tile staging via GLOBAL_LOAD_ASYNC_TO_LDS_B128 (ASYNCcnt), no VGPR
//     transit, no scratch spills; LDS double buffered, 1 barrier per K-tile
//   - sigmoid = v_exp_f32 + v_rcp_f32 (no IEEE divide sequence)
// ---------------------------------------------------------------------------

typedef __bf16 bf16_t;
typedef __attribute__((ext_vector_type(16))) __bf16 v16bf;
typedef __attribute__((ext_vector_type(8)))  float  v8f;
typedef __attribute__((ext_vector_type(4)))  int    v4i;

#define EP_DT   0.1f
#define EP_BETA 0.1f

#if __has_builtin(__builtin_amdgcn_global_load_async_to_lds_b128) && \
    __has_builtin(__builtin_amdgcn_s_wait_asynccnt)
#define HAVE_ASYNC_LDS 1
#else
#define HAVE_ASYNC_LDS 0
#endif

__device__ __forceinline__ float sigmoidf_(float x) {
  return __builtin_amdgcn_rcpf(1.0f + __expf(-x));
}

#if HAVE_ASYNC_LDS
typedef __attribute__((address_space(1))) v4i* g_v4i_p;
typedef __attribute__((address_space(3))) v4i* l_v4i_p;
__device__ __forceinline__ void async_b128(const void* g, void* l) {
  __builtin_amdgcn_global_load_async_to_lds_b128((g_v4i_p)g, (l_v4i_p)l, 0, 0);
}
#endif

__device__ __forceinline__ void wait_async_() {
#if HAVE_ASYNC_LDS
  __builtin_amdgcn_s_wait_asynccnt(0);
#endif
}

// Swizzled global weight layout: [k/32][n/16][lane(32)][slot(16)]
//   lane = (n&15) + (k&16), slot = k&15
__device__ __forceinline__ size_t swz_idx(int k, int n, int N) {
  int kt = k >> 5, nj = n >> 4;
  int dl = (n & 15) + (k & 16);
  int t  = k & 15;
  return (((size_t)kt * (N >> 4) + nj) * 32 + dl) * 16 + t;
}

// ------------------------- weight prep / init kernels ----------------------

__global__ void k_w0swz(const float* __restrict__ w0, bf16_t* __restrict__ d) {
  int i = blockIdx.x * blockDim.x + threadIdx.x;
  if (i < 1024 * 1024) {
    int k = i >> 10, n = i & 1023;
    d[swz_idx(k, n, 1024)] = (bf16_t)w0[i];
  }
}

__global__ void k_w1swz(const float* __restrict__ w1, bf16_t* __restrict__ d) {
  int i = blockIdx.x * blockDim.x + threadIdx.x;
  if (i < 1024 * 512) {
    int k = i >> 9, n = i & 511;
    d[swz_idx(k, n, 512)] = (bf16_t)w1[i];
  }
}

__global__ void k_w1tswz(const float* __restrict__ w1, bf16_t* __restrict__ d) {
  int i = blockIdx.x * blockDim.x + threadIdx.x;
  if (i < 512 * 1024) {
    int k = i >> 10, n = i & 1023;
    d[swz_idx(k, n, 1024)] = (bf16_t)w1[n * 512 + k];
  }
}

// W2 [512x10] N-padded to 16
__global__ void k_w2swz(const float* __restrict__ w2, bf16_t* __restrict__ d) {
  int i = blockIdx.x * blockDim.x + threadIdx.x;
  if (i < 512 * 16) {
    int k = i >> 4, n = i & 15;
    d[swz_idx(k, n, 16)] = (n < 10) ? (bf16_t)w2[k * 10 + n] : (bf16_t)0.0f;
  }
}

// W2^T K-padded to 32 [32x512]
__global__ void k_w2tswz(const float* __restrict__ w2, bf16_t* __restrict__ d) {
  int i = blockIdx.x * blockDim.x + threadIdx.x;
  if (i < 32 * 512) {
    int k = i >> 9, n = i & 511;
    d[swz_idx(k, n, 512)] = (k < 10) ? (bf16_t)w2[n * 10 + k] : (bf16_t)0.0f;
  }
}

// sigbx = sig(x) as bf16 (x is clamped -> once per call)
__global__ void k_sigx(const float* __restrict__ x, bf16_t* __restrict__ d) {
  int i = blockIdx.x * blockDim.x + threadIdx.x;
  if (i < 8192 * 1024) d[i] = (bf16_t)sigmoidf_(x[i]);
}

// states start at ones; sig-buffers start at sig(1); s3/sigb3 padded cols -> 0
__global__ void k_init(float* __restrict__ s1, float* __restrict__ s2, float* __restrict__ s3,
                       bf16_t* __restrict__ g1, bf16_t* __restrict__ g2, bf16_t* __restrict__ g3) {
  size_t i = (size_t)blockIdx.x * blockDim.x + threadIdx.x;
  const float sg1 = sigmoidf_(1.0f);
  if (i < 8192ull * 1024) { s1[i] = 1.0f; g1[i] = (bf16_t)sg1; }
  if (i < 8192ull * 512)  { s2[i] = 1.0f; g2[i] = (bf16_t)sg1; }
  if (i < 8192ull * 16)   { s3[i] = ((i & 15) < 10) ? 1.0f : 0.0f; }
  if (i < 8192ull * 32)   { g3[i] = ((i & 31) < 10) ? (bf16_t)sg1 : (bf16_t)0.0f; }
}

// ------------------------------ WMMA GEMM ----------------------------------
// MODE 0: FFX = sigbx @ W0 + b0
// MODE 1: s1 -= DT*( sig'(s1)*(FFX + sigb2@W1t) - s1 );          sigb1 = sig(s1')
// MODE 2: s2 -= DT*( sig'(s2)*(sigb1@W1 + sigb3@W2t + b1) - s2 ); sigb2 = sig(s2')
// MODE 3: s3 -= DT*( sig'(s3)*(sigb2@W2 + b2) - s3 + B*(oh-s3) ); sigb3 = sig(s3')

template <int MODE, int BM, int BN, int WAVES_M, int WAVES_N, int MI, int NI,
          int KTILES, int LDA, int LDC, int NSW, int LDSIG>
__global__ __launch_bounds__(WAVES_M * WAVES_N * 32)
void k_gemm(const bf16_t* __restrict__ A,       // sig(state) bf16, row-major
            const bf16_t* __restrict__ B,       // swizzled weights
            const float* __restrict__ FFX,
            const bf16_t* __restrict__ A2,      // MODE 2: sigb3 [8192x32], zero-padded
            const bf16_t* __restrict__ B2,      // MODE 2: W2t swizzled [1][32][512]
            const float* __restrict__ bias,
            float* __restrict__ C,
            bf16_t* __restrict__ SIG,           // sig(state_new) out (modes 1..3)
            const int* __restrict__ target) {
  constexpr int THREADS  = WAVES_M * WAVES_N * 32;
  constexpr int NT       = BN / 16;
  constexpr int BK       = 32;
  constexpr int LDA_PAD  = 40;                           // 80B rows, 16B aligned
  constexpr int AITERS   = (BM * BK / 8) / THREADS;      // b128 = 8 bf16
  constexpr int BVECS    = NT * 512 / 8;
  constexpr int BITERS   = (BVECS + THREADS - 1) / THREADS;
  constexpr int KT_TOTAL = KTILES + ((MODE == 2) ? 1 : 0);

  __shared__ __align__(16) bf16_t lA[2][BM * LDA_PAD];
  __shared__ __align__(16) bf16_t lB[2][NT * 512];

  const int tid  = threadIdx.x;
  const int lane = tid & 31;
  const int wid  = tid >> 5;
  const int wm   = wid / WAVES_N;
  const int wn   = wid % WAVES_N;
  const int half = lane >> 4;
  const int lrow = lane & 15;

  const int blockM = blockIdx.y * BM;
  const int blockN = blockIdx.x * BN;

  v8f acc[MI][NI];
#pragma unroll
  for (int mi = 0; mi < MI; ++mi)
#pragma unroll
    for (int ni = 0; ni < NI; ++ni)
#pragma unroll
      for (int e = 0; e < 8; ++e) acc[mi][ni][e] = 0.0f;

  // global tile -> LDS tile, b128 granularity, no VGPR data transit
  auto copy_tile = [&](int kt, int buf) {
    const bool extra = (MODE == 2) && (kt == KTILES);
    const bf16_t* Abase = extra ? A2 : (A + (size_t)kt * BK);
    const int astr = extra ? 32 : LDA;
#pragma unroll
    for (int i = 0; i < AITERS; ++i) {
      int v  = i * THREADS + tid;
      int r  = v >> 2;                 // 4 b128 per 32-wide row
      int cv = (v & 3) << 3;
      const bf16_t* g = Abase + (size_t)(blockM + r) * astr + cv;
      bf16_t* l = &lA[buf][r * LDA_PAD + cv];
#if HAVE_ASYNC_LDS
      async_b128(g, l);
#else
      *(uint4*)l = *(const uint4*)g;
#endif
    }
    const bf16_t* Bs = extra ? (B2 + (size_t)(blockN >> 4) * 512)
                             : (B + ((size_t)kt * NSW + (blockN >> 4)) * 512);
#pragma unroll
    for (int i = 0; i < BITERS; ++i) {
      int idx = i * THREADS + tid;
      if ((BVECS % THREADS) == 0 || idx < BVECS) {
#if HAVE_ASYNC_LDS
        async_b128((const uint4*)Bs + idx, (uint4*)lB[buf] + idx);
#else
        ((uint4*)lB[buf])[idx] = ((const uint4*)Bs)[idx];
#endif
      }
    }
    if (!extra && kt + 1 < KTILES)
      __builtin_prefetch(B + ((size_t)(kt + 1) * NSW + (blockN >> 4)) * 512, 0, 1);
  };

  auto mma_tiles = [&](int buf) {
    // hoist B fragments (NI of them), stream A fragments one at a time
    v16bf bfr[NI];
#pragma unroll
    for (int ni = 0; ni < NI; ++ni) {
      const bf16_t* p = &lB[buf][(wn * NI + ni) * 512 + lane * 16];
      uint4* d = (uint4*)&bfr[ni];
      d[0] = *(const uint4*)p;
      d[1] = *(const uint4*)(p + 8);
    }
#pragma unroll
    for (int mi = 0; mi < MI; ++mi) {
      // A 16x32 bf16 frag: lane<16 -> K{0..15}; lane>=16 -> K{16..31}
      v16bf af;
      const bf16_t* p = &lA[buf][(wm * MI * 16 + mi * 16 + lrow) * LDA_PAD + half * 16];
      uint4* d = (uint4*)&af;
      d[0] = *(const uint4*)p;
      d[1] = *(const uint4*)(p + 8);
#pragma unroll
      for (int ni = 0; ni < NI; ++ni)
        acc[mi][ni] = __builtin_amdgcn_wmma_f32_16x16x32_bf16(
            false, af, false, bfr[ni], (short)0, acc[mi][ni], false, false);
    }
  };

  // software-pipelined main loop, one barrier per K-tile
  copy_tile(0, 0);
  wait_async_();
  __syncthreads();
  for (int kt = 0; kt < KT_TOTAL; ++kt) {
    int buf = kt & 1;
    bool haveNext = (kt + 1 < KT_TOTAL);
    if (haveNext) copy_tile(kt + 1, buf ^ 1);   // async copy overlaps WMMA below
    mma_tiles(buf);
    if (haveNext) wait_async_();
    __syncthreads();
  }

  // epilogue: accumulator element e -> (m = e + 8*half, n = lane&15)
#pragma unroll
  for (int mi = 0; mi < MI; ++mi)
#pragma unroll
    for (int ni = 0; ni < NI; ++ni)
#pragma unroll
      for (int e = 0; e < 8; ++e) {
        int row = blockM + wm * MI * 16 + mi * 16 + e + 8 * half;
        int col = blockN + wn * NI * 16 + ni * 16 + lrow;
        float a = acc[mi][ni][e];
        if constexpr (MODE == 0) {
          C[(size_t)row * LDC + col] = a + bias[col];
        } else if constexpr (MODE == 1) {
          size_t idx = (size_t)row * LDC + col;
          float s = C[idx];
          float si = sigmoidf_(s);
          float grad = si * (1.0f - si) * (FFX[idx] + a) - s;
          float sn = s - EP_DT * grad;
          C[idx] = sn;
          SIG[(size_t)row * LDSIG + col] = (bf16_t)sigmoidf_(sn);
        } else if constexpr (MODE == 2) {
          size_t idx = (size_t)row * LDC + col;
          float s = C[idx];
          float si = sigmoidf_(s);
          float grad = si * (1.0f - si) * (a + bias[col]) - s;
          float sn = s - EP_DT * grad;
          C[idx] = sn;
          SIG[(size_t)row * LDSIG + col] = (bf16_t)sigmoidf_(sn);
        } else {
          size_t idx = (size_t)row * LDC + col;
          size_t gix = (size_t)row * LDSIG + col;
          if (col < 10) {
            float s = C[idx];
            float sl = sigmoidf_(s);
            float grad = sl * (1.0f - sl) * (a + bias[col]) - s;
            float oh = (col == target[row]) ? 1.0f : 0.0f;
            grad += EP_BETA * (oh - s);
            float sn = s - EP_DT * grad;
            C[idx] = sn;
            SIG[gix] = (bf16_t)sigmoidf_(sn);
          } else {
            C[idx] = 0.0f;
            SIG[gix] = (bf16_t)0.0f;
          }
        }
      }
}

// out[b, :] = concat(x[b], s1[b], s2[b], s3[b, 0:10])  -> [8192, 2570]
__global__ void k_pack(const float* __restrict__ x, const float* __restrict__ s1,
                       const float* __restrict__ s2, const float* __restrict__ s3,
                       float* __restrict__ out) {
  size_t i = (size_t)blockIdx.x * blockDim.x + threadIdx.x;
  const size_t total = 8192ull * 2570ull;
  if (i >= total) return;
  size_t b = i / 2570;
  int j = (int)(i % 2570);
  float v;
  if (j < 1024)      v = x [b * 1024 + j];
  else if (j < 2048) v = s1[b * 1024 + (j - 1024)];
  else if (j < 2560) v = s2[b * 512  + (j - 2048)];
  else               v = s3[b * 16   + (j - 2560)];
  out[i] = v;
}

// ------------------------------- launcher ----------------------------------

extern "C" void kernel_launch(void* const* d_in, const int* in_sizes, int n_in,
                              void* d_out, int out_size, void* d_ws, size_t ws_size,
                              hipStream_t stream) {
  (void)in_sizes; (void)n_in; (void)out_size; (void)ws_size;

  const float* x  = (const float*)d_in[0];
  const int*   tg = (const int*)d_in[1];
  const float* W0 = (const float*)d_in[2];
  const float* W1 = (const float*)d_in[3];
  const float* W2 = (const float*)d_in[4];
  const float* b0 = (const float*)d_in[5];
  const float* b1 = (const float*)d_in[6];
  const float* b2 = (const float*)d_in[7];

  char* ws = (char*)d_ws;
  auto take = [&](size_t bytes) -> char* {
    char* p = ws;
    ws += (bytes + 255) & ~(size_t)255;
    return p;
  };
  float*  FFX  = (float*)take(8192ull * 1024 * 4);  // sig(x)@W0 + b0 (loop invariant)
  float*  s1   = (float*)take(8192ull * 1024 * 4);
  float*  s2   = (float*)take(8192ull * 512 * 4);
  float*  s3   = (float*)take(8192ull * 16 * 4);    // N padded 10 -> 16
  bf16_t* gX   = (bf16_t*)take(8192ull * 1024 * 2); // sig(x)
  bf16_t* g1   = (bf16_t*)take(8192ull * 1024 * 2); // sig(s1)
  bf16_t* g2   = (bf16_t*)take(8192ull * 512 * 2);  // sig(s2)
  bf16_t* g3   = (bf16_t*)take(8192ull * 32 * 2);   // sig(s3), 32-wide zero-padded
  bf16_t* W0b  = (bf16_t*)take(1024ull * 1024 * 2); // all weight copies swizzled
  bf16_t* W1b  = (bf16_t*)take(1024ull * 512 * 2);
  bf16_t* W1t  = (bf16_t*)take(512ull * 1024 * 2);
  bf16_t* W2b  = (bf16_t*)take(512ull * 16 * 2);
  bf16_t* W2t  = (bf16_t*)take(32ull * 512 * 2);

  dim3 blk(256);
  k_w0swz <<<(1024 * 1024 + 255) / 256, blk, 0, stream>>>(W0, W0b);
  k_w1swz <<<(1024 * 512 + 255) / 256, blk, 0, stream>>>(W1, W1b);
  k_w1tswz<<<(512 * 1024 + 255) / 256, blk, 0, stream>>>(W1, W1t);
  k_w2swz <<<(512 * 16 + 255) / 256, blk, 0, stream>>>(W2, W2b);
  k_w2tswz<<<(32 * 512 + 255) / 256, blk, 0, stream>>>(W2, W2t);
  k_sigx  <<<(8192 * 1024 + 255) / 256, blk, 0, stream>>>(x, gX);
  k_init  <<<(8192 * 1024 + 255) / 256, blk, 0, stream>>>(s1, s2, s3, g1, g2, g3);

  // FFX = sig(x) @ W0 + b0   (once; x is clamped)
  {
    dim3 g(1024 / 128, 8192 / 128);
    k_gemm<0, 128, 128, 2, 4, 4, 2, 32, 1024, 1024, 64, 0><<<g, blk, 0, stream>>>(
        gX, W0b, nullptr, nullptr, nullptr, b0, FFX, nullptr, nullptr);
  }

  for (int t = 0; t < 20; ++t) {
    {  // s1 update: fb = sig(s2) @ W1^T
      dim3 g(1024 / 128, 8192 / 128);
      k_gemm<1, 128, 128, 2, 4, 4, 2, 16, 512, 1024, 64, 1024><<<g, blk, 0, stream>>>(
          g2, W1t, FFX, nullptr, nullptr, nullptr, s1, g1, nullptr);
    }
    {  // s2 update: sig(s1_new) @ W1  (+ fused sig(s3) @ W2^T extra K-tile)
      dim3 g(512 / 128, 8192 / 128);
      k_gemm<2, 128, 128, 2, 4, 4, 2, 32, 1024, 512, 32, 512><<<g, blk, 0, stream>>>(
          g1, W1b, nullptr, g3, W2t, b1, s2, g2, nullptr);
    }
    {  // s3 update: sig(s2_new) @ W2 + beta*(onehot - s3)
      dim3 g(1, 8192 / 128);
      k_gemm<3, 128, 16, 8, 1, 1, 1, 16, 512, 16, 1, 32><<<g, blk, 0, stream>>>(
          g2, W2b, nullptr, nullptr, nullptr, b2, s3, g3, tg);
    }
  }

  size_t total = 8192ull * 2570ull;
  k_pack<<<(unsigned)((total + 255) / 256), blk, 0, stream>>>(x, s1, s2, s3, (float*)d_out);
}
